// GaussianWeightedSE_51230369906981
// MI455X (gfx1250) — compile-verified, matrix operand order checked by reading, not run on hardware
//
#include <hip/hip_runtime.h>
#include <hip/hip_bf16.h>

typedef __attribute__((ext_vector_type(16))) _Float16 v16h;
typedef __attribute__((ext_vector_type(8)))  float    v8f;

// ---------------------------------------------------------------------------
// Kernel 1: per-batch normalized Gaussian kernel -> knorm[b][4096]
// ---------------------------------------------------------------------------
__global__ void gw_gauss(const float* __restrict__ mu,
                         const float* __restrict__ log_sigma,
                         float* __restrict__ knorm) {
    const int b = blockIdx.x;                 // 32 blocks, 256 threads
    __shared__ float red[256];
    const float sigma = __expf(log_sigma[0]);
    const float inv2s = 1.0f / (2.0f * sigma);
    const float muh = mu[b * 2 + 0];
    const float muw = mu[b * 2 + 1];
    float* kb = knorm + b * 4096;

    float acc = 0.0f;
    for (int i = threadIdx.x; i < 4096; i += 256) {
        const int h = i >> 6, w = i & 63;
        const float dx = (float)h * (1.0f / 63.0f) - muh;
        const float dy = (float)w * (1.0f / 63.0f) - muw;
        const float v = __expf(-(dx * dx + dy * dy) * inv2s);
        kb[i] = v;
        acc += v;
    }
    red[threadIdx.x] = acc;
    __syncthreads();
    for (int s = 128; s > 0; s >>= 1) {
        if ((int)threadIdx.x < s) red[threadIdx.x] += red[threadIdx.x + s];
        __syncthreads();
    }
    const float inv = 1.0f / red[0];
    for (int i = threadIdx.x; i < 4096; i += 256) kb[i] *= inv;
}

// ---------------------------------------------------------------------------
// Kernel 2: squeezed[b][c] = sum_i x[b,c,i] * knorm[b,i]
// grid = 32*64 blocks, 256 threads (8 waves); one wave per channel.
// Kernel tile staged in LDS once per block (reused by 8 waves).
// ---------------------------------------------------------------------------
__global__ void gw_pool(const float* __restrict__ x,
                        const float* __restrict__ knorm,
                        float* __restrict__ squeezed) {
    const int b  = blockIdx.x >> 6;
    const int cg = blockIdx.x & 63;
    __shared__ float4 kch[1024];

    const float4* kb = (const float4*)(knorm + b * 4096);
    for (int i = threadIdx.x; i < 1024; i += 256) kch[i] = kb[i];
    __syncthreads();

    const int wave = threadIdx.x >> 5;
    const int lane = threadIdx.x & 31;
    const int c = cg * 8 + wave;

    const float4* xr = (const float4*)(x + ((size_t)(b * 512 + c)) * 4096);
    float acc = 0.0f;
    for (int i = lane; i < 1024; i += 32) {   // coalesced 512B per wave access
        const float4 xv = xr[i];
        const float4 kv = kch[i];
        acc += xv.x * kv.x + xv.y * kv.y + xv.z * kv.z + xv.w * kv.w;
    }
    // wave32 reduction
    for (int off = 16; off > 0; off >>= 1) acc += __shfl_down(acc, off, 32);
    if (lane == 0) squeezed[b * 512 + c] = acc;
}

// ---------------------------------------------------------------------------
// Kernel 3: SE-MLP entirely in WMMA.
//   hid     = relu(squeezed(32x512) @ w_fc1^T(512x32))      -> LDS
//   weights = sigmoid(hid(32x32)    @ w_fc2^T(32x512))      -> d_ws
// 1 block x 128 threads (4 waves). ISA 16-bit A-layout packing:
//   lane L (0..15): M=L, halves 0..7 -> K=kh..kh+7, halves 8..15 -> K=kh+16..kh+23, kh=0
//   lane L (16..31): same rows, kh=8.
// B for (X @ W^T) has identical packing applied to row-major W (n<->m symmetry).
// ---------------------------------------------------------------------------
__global__ void gw_se_mlp(const float* __restrict__ squeezed,
                          const float* __restrict__ w_fc1,
                          const float* __restrict__ w_fc2,
                          float* __restrict__ weights) {
    __shared__ float hid[32 * 32];
    const int wave = threadIdx.x >> 5;
    const int lane = threadIdx.x & 31;
    const int r  = lane & 15;           // row (A) / col (B) within tile
    const int kh = (lane >> 4) << 3;    // 0 or 8

    // ---- phase 1: hid tiles, one 16x16 tile per wave, K-loop over 512 ----
    {
        const int m0 = (wave >> 1) << 4;
        const int n0 = (wave & 1) << 4;
        v8f c = {};
        for (int k0 = 0; k0 < 512; k0 += 32) {
            const float* arow = squeezed + (m0 + r) * 512 + k0 + kh;
            const float* brow = w_fc1    + (n0 + r) * 512 + k0 + kh;
            v16h a, bf;
#pragma unroll
            for (int j = 0; j < 8; ++j) {
                a[j]      = (_Float16)arow[j];
                a[8 + j]  = (_Float16)arow[16 + j];
                bf[j]     = (_Float16)brow[j];
                bf[8 + j] = (_Float16)brow[16 + j];
            }
            c = __builtin_amdgcn_wmma_f32_16x16x32_f16(
                    false, a, false, bf, (short)0, c, false, false);
        }
#pragma unroll
        for (int g = 0; g < 8; ++g) {
            const float v = c[g] > 0.0f ? c[g] : 0.0f;      // relu
            const int row = m0 + ((lane >> 4) << 3) + g;
            hid[row * 32 + n0 + r] = v;
        }
    }
    __syncthreads();

    // ---- phase 2: 64 tiles of weights (2 M-tiles x 32 N-tiles), K=32 ----
    for (int t = wave; t < 64; t += 4) {
        const int m0 = (t & 1) << 4;
        const int n0 = (t >> 1) << 4;
        const float* arow = hid   + (m0 + r) * 32 + kh;
        const float* brow = w_fc2 + (n0 + r) * 32 + kh;
        v16h a, bf;
#pragma unroll
        for (int j = 0; j < 8; ++j) {
            a[j]      = (_Float16)arow[j];
            a[8 + j]  = (_Float16)arow[16 + j];
            bf[j]     = (_Float16)brow[j];
            bf[8 + j] = (_Float16)brow[16 + j];
        }
        v8f c = {};
        c = __builtin_amdgcn_wmma_f32_16x16x32_f16(
                false, a, false, bf, (short)0, c, false, false);
#pragma unroll
        for (int g = 0; g < 8; ++g) {
            const float v = 1.0f / (1.0f + __expf(-c[g]));  // sigmoid
            const int row = m0 + ((lane >> 4) << 3) + g;
            weights[row * 512 + n0 + r] = v;
        }
    }
}

// ---------------------------------------------------------------------------
// Kernel 4: out[b,c,:,:] = x[b,c,:,:] * weights[b,c]   (float4 streaming)
// ---------------------------------------------------------------------------
__global__ void gw_scale(const float* __restrict__ x,
                         const float* __restrict__ weights,
                         float* __restrict__ out) {
    const size_t i = (size_t)blockIdx.x * blockDim.x + threadIdx.x; // float4 idx
    const float w = weights[i >> 10];        // 1024 float4 per (b,c)
    const float4 v = ((const float4*)x)[i];
    float4 o;
    o.x = v.x * w; o.y = v.y * w; o.z = v.z * w; o.w = v.w * w;
    ((float4*)out)[i] = o;
}

// ---------------------------------------------------------------------------
extern "C" void kernel_launch(void* const* d_in, const int* in_sizes, int n_in,
                              void* d_out, int out_size, void* d_ws, size_t ws_size,
                              hipStream_t stream) {
    const float* x         = (const float*)d_in[0];   // (32,512,64,64)
    const float* mu        = (const float*)d_in[1];   // (32,2)
    const float* log_sigma = (const float*)d_in[2];   // scalar
    const float* w_fc1     = (const float*)d_in[3];   // (32,512)
    const float* w_fc2     = (const float*)d_in[4];   // (512,32)
    float* out             = (float*)d_out;

    // workspace layout
    char* ws = (char*)d_ws;
    float* knorm    = (float*)(ws);                       // 32*4096 f32 = 512 KB
    float* squeezed = (float*)(ws + 512 * 1024);          // 32*512  f32 =  64 KB
    float* weights  = (float*)(ws + 512 * 1024 + 64 * 1024); // 32*512 f32 = 64 KB

    gw_gauss<<<32, 256, 0, stream>>>(mu, log_sigma, knorm);
    gw_pool<<<32 * 64, 256, 0, stream>>>(x, knorm, squeezed);
    gw_se_mlp<<<1, 128, 0, stream>>>(squeezed, w_fc1, w_fc2, weights);
    gw_scale<<<65536, 256, 0, stream>>>(x, weights, out);  // 16,777,216 float4
}
